// ruud_MPQE_39668317946545
// MI455X (gfx1250) — matrix operand order repaired and checked by reading, not compiled, not run on hardware
//
#include <hip/hip_runtime.h>
#include <hip/hip_bf16.h>
#include <stdint.h>

// ---------------------------------------------------------------------------
// MPQE '3i' RGCN (3 layers, basis-decomposed) for MI455X / gfx1250.
//
//  * Regular '3i' topology: anchors {4q,4q+1,4q+2} -> target 4q+3. No scatter,
//    mean divisor == 3, anchors receive zero aggregation.
//  * Basis trick: sum_k c_kb*(a_k@basis_b) = sum_b (sum_k c_kb a_k)@basis_b
//    -> every GEMM uses one of the 8 *shared* basis matrices => M=16 WMMA
//       tiles; comp coefficients are applied per-lane on the A side.
//  * Weights pre-converted once to bf16 [N][K] (transposed, padded) in d_ws;
//    the main kernel DOUBLE-BUFFERS them into LDS with async global->LDS DMA
//    (ASYNCcnt) overlapped with the WMMA work on the previous matrix.
// ---------------------------------------------------------------------------

#define D      128
#define L_NUM  3
#define NB     8
#define R_NUM  100
#define BQ     65536
#define PITCH  136                 // bf16 elems per row (128 + 8 pad)
#define WAVES  8
#define QPW    16                  // queries (M rows) per wave
#define WMAT_U16 (D * PITCH)       // one staged weight matrix, in ushort elems
#define WMAT_BYTES (WMAT_U16 * 2)  // 34816 B

typedef __bf16 bf16_t;
typedef bf16_t v16bf __attribute__((ext_vector_type(16)));
typedef float  v8f   __attribute__((ext_vector_type(8)));

union FragU { uint4 u[2]; v16bf v; };

static __device__ __forceinline__ float bf2f(unsigned int u) {
  union { unsigned int i; float f; } x; x.i = u << 16; return x.f;
}
static __device__ __forceinline__ unsigned short f2bf(float f) {
  union { float f; unsigned int i; } x; x.f = f;
  unsigned int i = x.i;
  return (unsigned short)((i + 0x7FFFu + ((i >> 16) & 1u)) >> 16);  // RNE
}
// packed f32x2 -> bf16x2 (native v_cvt_pk_bf16_f32 when available)
#if __has_builtin(__builtin_amdgcn_cvt_pk_bf16_f32)
typedef bf16_t v2bf __attribute__((ext_vector_type(2)));
static __device__ __forceinline__ unsigned pack2bf(float lo, float hi) {
  union { v2bf v; unsigned u; } x;
  x.v = __builtin_amdgcn_cvt_pk_bf16_f32(lo, hi);
  return x.u;
}
#else
static __device__ __forceinline__ unsigned pack2bf(float lo, float hi) {
  return (unsigned)f2bf(lo) | ((unsigned)f2bf(hi) << 16);
}
#endif
static __device__ __forceinline__ v8f vzero() {
  v8f v = {0.f, 0.f, 0.f, 0.f, 0.f, 0.f, 0.f, 0.f};
  return v;
}

// A fragment (16x32 bf16), rows 0..15 of a row-major bf16 region.
// lane<16: M=lane, K=kbase+{0..7},{16..23}; lane>=16: M=lane-16, K=+{8..15},{24..31}
static __device__ __forceinline__ v16bf load_a(const unsigned short* base, int lane, int kbase) {
  int row = lane & 15;
  int k0  = kbase + ((lane >> 4) << 3);
  const unsigned short* p = base + row * PITCH + k0;
  FragU f;
  f.u[0] = *(const uint4*)(p);
  f.u[1] = *(const uint4*)(p + 16);
  return f.v;
}

// B fragment (32x16 bf16) from [N][K] staged weight.
// lane<16: N=nbase+lane, K=kbase+{0..7},{16..23}; lane>=16: N=nbase+lane-16, K=kbase+{8..15},{24..31}
static __device__ __forceinline__ v16bf load_b(const unsigned short* wbase, int lane,
                                               int kbase, int nbase) {
  int col = nbase + (lane & 15);
  int k0  = kbase + ((lane >> 4) << 3);
  const unsigned short* p = wbase + col * PITCH + k0;
  FragU f;
  f.u[0] = *(const uint4*)(p);
  f.u[1] = *(const uint4*)(p + 16);
  return f.v;
}

// 16x128 tile @ 128x128 staged weight -> acc[8].  B-fragments for a whole
// K-slice are batched into registers first so the 8 WMMAs issue back-to-back.
static __device__ __forceinline__ void gemm_tile(const unsigned short* abase,
                                                 const unsigned short* wbase,
                                                 int lane, v8f acc[8]) {
#pragma unroll
  for (int kf = 0; kf < 4; ++kf) {
    v16bf a = load_a(abase, lane, kf * 32);
    v16bf bf[8];
#pragma unroll
    for (int nf = 0; nf < 8; ++nf) bf[nf] = load_b(wbase, lane, kf * 32, nf * 16);
#pragma unroll
    for (int nf = 0; nf < 8; ++nf)
      acc[nf] = __builtin_amdgcn_wmma_f32_16x16x32_bf16(
          false, a, false, bf[nf], (short)0, acc[nf], false, false);
  }
}

// async DMA of one 34KB bf16 weight matrix: global -> LDS (no VGPR round trip,
// tracked by ASYNCcnt).  dst_lds = 32-bit LDS byte address of the buffer.
static __device__ __forceinline__ void stage_async(unsigned dst_lds,
                                                   const unsigned short* src, int tid) {
  unsigned long long base = (unsigned long long)(uintptr_t)src;
  for (int i = tid; i < WMAT_BYTES / 16; i += 256) {
    unsigned lds_addr = dst_lds + (unsigned)i * 16u;
    unsigned voff     = (unsigned)i * 16u;
    asm volatile("global_load_async_to_lds_b128 %0, %1, %2"
                 :: "v"(lds_addr), "v"(voff), "s"(base) : "memory");
  }
}
static __device__ __forceinline__ void wait_async() {
  asm volatile("s_wait_asynccnt 0x0" ::: "memory");
}

// matrix schedule: per layer, 8 basis matrices then the root matrix
static __device__ __forceinline__ const unsigned short* mat_ptr(const unsigned short* wt, int s) {
  int l = s / 9, step = s - l * 9;
  int m = (step < 8) ? (l * NB + step) : (24 + l);
  return wt + (size_t)m * WMAT_U16;
}

// ---------------------------------------------------------------------------
// Kernel 1: fp32 basis/root -> bf16, transposed to [N][K] with padded pitch.
// ---------------------------------------------------------------------------
__global__ void mpqe_prep_weights(const float* __restrict__ basis,
                                  const float* __restrict__ root,
                                  unsigned short* __restrict__ wt) {
  int t = blockIdx.x * blockDim.x + threadIdx.x;
  if (t >= 27 * D) return;
  int m = t / D, n = t % D;
  const float* src = (m < 24) ? (basis + (size_t)m * D * D)
                              : (root + (size_t)(m - 24) * D * D);
  unsigned short* dst = wt + (size_t)m * WMAT_U16 + (size_t)n * PITCH;
  for (int k = 0; k < D; ++k) dst[k] = f2bf(src[(size_t)k * D + n]);  // transpose
  for (int k = D; k < PITCH; ++k) dst[k] = 0;                         // pad
}

// ---------------------------------------------------------------------------
// Kernel 2: fused 3-layer RGCN + target readout. One wave = 16 queries.
// ---------------------------------------------------------------------------
__global__ void __launch_bounds__(256)
mpqe_rgcn3i_kernel(const float* __restrict__ x,
                   const float* __restrict__ comp,
                   const float* __restrict__ bias,
                   const int*   __restrict__ edge_type,
                   const unsigned short* __restrict__ wt,
                   float* __restrict__ out) {
  extern __shared__ char smem[];
  unsigned short* lds = (unsigned short*)smem;
  const int tid  = threadIdx.x;
  const int wave = tid >> 5;
  const int lane = tid & 31;

  unsigned short* state = lds + wave * (64 * PITCH);                       // row = node*16+q
  unsigned short* sreg  = lds + WAVES * 64 * PITCH + wave * (QPW * PITCH); // S_b rows
  unsigned short* wb0   = lds + WAVES * 64 * PITCH + WAVES * QPW * PITCH;  // 2 weight bufs
  // 32-bit LDS byte address of wb0 for the async DMA (flat LDS addr [31:0]);
  // single cast of the base keeps the ds-load pointer chains inferable.
  const unsigned wb0_lds =
      (unsigned)(uintptr_t)lds + (unsigned)((WAVES * 64 * PITCH + WAVES * QPW * PITCH) * 2);

  const int qbase = blockIdx.x * (WAVES * QPW) + wave * QPW;

  // prologue: start DMA of the first basis matrix, overlapped with x load
  stage_async(wb0_lds, mat_ptr(wt, 0), tid);

  // ---- load x rows into per-wave state (fp32 -> bf16): float4 in, uint2 out
  for (int r = 0; r < 64; ++r) {
    long g = ((long)(qbase + (r & 15)) * 4 + (r >> 4)) * D;
    float4 v = *(const float4*)(x + g + lane * 4);
    uint2 pk;
    pk.x = pack2bf(v.x, v.y);
    pk.y = pack2bf(v.z, v.w);
    *(uint2*)(state + r * PITCH + lane * 4) = pk;
  }

  // per-lane query row + its 3 edge relation types (lane and lane+16 share q)
  const int q    = lane & 15;
  const int half = lane >> 4;
  const int rt0  = edge_type[(qbase + q) * 3 + 0];
  const int rt1  = edge_type[(qbase + q) * 3 + 1];
  const int rt2  = edge_type[(qbase + q) * 3 + 2];

  wait_async();
  __syncthreads();
  int cur = 0;

  const int col_lo = lane & 15;
  const int rbase  = half ? 8 : 0;

  for (int l = 0; l < L_NUM; ++l) {
    v8f macc[8];
#pragma unroll
    for (int nf = 0; nf < 8; ++nf) macc[nf] = vzero();

    for (int step = 0; step < 9; ++step) {
      const int s = l * 9 + step;
      if (s + 1 < 27)                                     // prefetch next matrix
        stage_async(wb0_lds + (unsigned)(cur ^ 1) * (unsigned)WMAT_BYTES,
                    mat_ptr(wt, s + 1), tid);

      const unsigned short* wcur = wb0 + (unsigned)cur * (unsigned)WMAT_U16;

      if (step < 8) {
        const int b = step;
        { // build S_b rows (per-wave, per-lane covers half a row, 2 cols/iter)
          const float s3 = 1.0f / 3.0f;                   // fold mean divisor
          float c0 = comp[(size_t)(l * R_NUM + rt0) * NB + b] * s3;
          float c1 = comp[(size_t)(l * R_NUM + rt1) * NB + b] * s3;
          float c2 = comp[(size_t)(l * R_NUM + rt2) * NB + b] * s3;
          const unsigned int* p0 = (const unsigned int*)(state + (0 * 16 + q) * PITCH + half * 64);
          const unsigned int* p1 = (const unsigned int*)(state + (1 * 16 + q) * PITCH + half * 64);
          const unsigned int* p2 = (const unsigned int*)(state + (2 * 16 + q) * PITCH + half * 64);
          unsigned int* sp = (unsigned int*)(sreg + q * PITCH + half * 64);
          for (int c = 0; c < 32; ++c) {
            unsigned int w0 = p0[c], w1 = p1[c], w2 = p2[c];
            float lo = c0 * bf2f(w0 & 0xFFFFu) + c1 * bf2f(w1 & 0xFFFFu) + c2 * bf2f(w2 & 0xFFFFu);
            float hi = c0 * bf2f(w0 >> 16)     + c1 * bf2f(w1 >> 16)     + c2 * bf2f(w2 >> 16);
            sp[c] = pack2bf(lo, hi);
          }
        }
        gemm_tile(sreg, wcur, lane, macc);                // 32 x v_wmma bf16
      } else {
        // ---- root step: target @ root accumulates into macc
        gemm_tile(state + 3 * 16 * PITCH, wcur, lane, macc);

        if (l < L_NUM - 1) {
          // new target state (bias + relu) -> LDS, freeing macc before anchors
#pragma unroll
          for (int nf = 0; nf < 8; ++nf) {
            float bv = bias[l * D + nf * 16 + col_lo];
#pragma unroll
            for (int j = 0; j < 8; ++j) {
              float v = macc[nf][j] + bv;
              v = v > 0.f ? v : 0.f;
              state[(48 + rbase + j) * PITCH + nf * 16 + col_lo] = f2bf(v);
            }
          }
          // anchors @ root -> new anchor states (bias + relu)
          for (int k = 0; k < 3; ++k) {
            v8f aacc[8];
#pragma unroll
            for (int nf = 0; nf < 8; ++nf) aacc[nf] = vzero();
            gemm_tile(state + k * 16 * PITCH, wcur, lane, aacc);
#pragma unroll
            for (int nf = 0; nf < 8; ++nf) {
              float bv = bias[l * D + nf * 16 + col_lo];
#pragma unroll
              for (int j = 0; j < 8; ++j) {
                float v = aacc[nf][j] + bv;
                v = v > 0.f ? v : 0.f;
                state[(k * 16 + rbase + j) * PITCH + nf * 16 + col_lo] = f2bf(v);
              }
            }
          }
        } else {
          // final layer: write target rows straight to global out (f32, +bias)
#pragma unroll
          for (int nf = 0; nf < 8; ++nf) {
            float bv = bias[l * D + nf * 16 + col_lo];
#pragma unroll
            for (int j = 0; j < 8; ++j) {
              out[(size_t)(qbase + rbase + j) * D + nf * 16 + col_lo] = macc[nf][j] + bv;
            }
          }
        }
      }

      if (s + 1 < 27) {          // retire this step's DMA, publish buffer, flip
        wait_async();
        __syncthreads();
        cur ^= 1;
      }
    }
  }
}

// ---------------------------------------------------------------------------
extern "C" void kernel_launch(void* const* d_in, const int* in_sizes, int n_in,
                              void* d_out, int out_size, void* d_ws, size_t ws_size,
                              hipStream_t stream) {
  (void)in_sizes; (void)n_in; (void)out_size; (void)ws_size;
  const float* x         = (const float*)d_in[0];
  const float* basis     = (const float*)d_in[1];
  const float* comp      = (const float*)d_in[2];
  const float* root      = (const float*)d_in[3];
  const float* bias      = (const float*)d_in[4];
  /* d_in[5] edge_index: structure is regular/implicit for '3i' queries */
  const int*   edge_type = (const int*)d_in[6];
  float* out = (float*)d_out;
  unsigned short* wt = (unsigned short*)d_ws;   // 27 * 34816 B = ~0.94 MB scratch

  // 1) bf16/transpose all relation-basis + root matrices once
  mpqe_prep_weights<<<(27 * D + 255) / 256, 256, 0, stream>>>(basis, root, wt);

  // 2) main fused kernel: 512 blocks x 256 threads, 238KB dynamic LDS
  const int LDS_BYTES = (WAVES * 64 * PITCH + WAVES * QPW * PITCH) * 2 + 2 * WMAT_BYTES;
  hipFuncSetAttribute((const void*)mpqe_rgcn3i_kernel,
                      hipFuncAttributeMaxDynamicSharedMemorySize, LDS_BYTES);
  mpqe_rgcn3i_kernel<<<BQ / (WAVES * QPW), 256, LDS_BYTES, stream>>>(
      x, comp, bias, edge_type, wt, out);
}
